// DeepGRU_36756330120081
// MI455X (gfx1250) — compile-verified, matrix operand order checked by reading, not run on hardware
//
#include <hip/hip_runtime.h>
#include <hip/hip_bf16.h>

#define T_LEN 256
#define BATCH 32
#define HIDN  100
#define EMBD  50
#define VOC   32000
#define NFC   1019              // FC consumer blocks; grid = 5 + NFC = 1024
#define NJOBS (T_LEN * 50)      // FC tile jobs: t (256) x n-block (50 x 640 cols)

typedef __bf16 v16bf __attribute__((ext_vector_type(16)));
typedef float  v8f   __attribute__((ext_vector_type(8)));

union Frag { v16bf v; uint4 q[2]; };

static __device__ __forceinline__ unsigned short f2bf(float f) {
  union { float f; unsigned u; } c; c.f = f;
  unsigned u = c.u;
  unsigned r = u + 0x7FFFu + ((u >> 16) & 1u);   // round-to-nearest-even
  return (unsigned short)(r >> 16);
}

static __device__ __forceinline__ v8f wmma_bf16(v16bf a, v16bf b, v8f c) {
  return __builtin_amdgcn_wmma_f32_16x16x32_bf16(false, a, false, b, (short)0, c,
                                                 false, false);
}

// A fragment (16x32 bf16 tile) from a [rows][128] bf16 array.
// CDNA5 16-bit A layout: lanes 0-15 hold M rows with K={0..7,16..23},
// lanes 16-31 hold K={8..15,24..31}.
static __device__ __forceinline__ v16bf load_afrag(const unsigned short* base,
                                                   int m0, int kt, int lane) {
  int m  = m0 + (lane & 15);
  int kb = kt * 32 + ((lane >> 4) << 3);
  const unsigned short* p = base + m * 128 + kb;
  Frag f;
  f.q[0] = *(const uint4*)(p);
  f.q[1] = *(const uint4*)(p + 16);
  return f.v;
}

// B fragment from pre-packed buffer: 512 halves per fragment, per-lane 32B chunk.
static __device__ __forceinline__ v16bf load_bfrag(const unsigned short* packBase,
                                                   int fragIdx, int lane) {
  const unsigned short* p = packBase + fragIdx * 512 + lane * 16;
  Frag f;
  f.q[0] = *(const uint4*)(p);
  f.q[1] = *(const uint4*)(p + 8);
  return f.v;
}

// ---------------------------------------------------------------------------
// Weight packing: f32 row-major -> bf16 WMMA-B fragments.
// Fragment element (lane, h): n = nt*16 + (lane&15),
//                             k = kt*32 + (lane>>4)*8 + (h<8 ? h : h+8)
// ---------------------------------------------------------------------------
struct PackArgs { const float* w[30]; };

__global__ void pack_gru_kernel(PackArgs args, unsigned short* wpack) {
  const int perMat = 4 * 7 * 512;  // 14336 halves per 128x112 matrix
  const int total  = 30 * perMat;
  for (int idx = blockIdx.x * blockDim.x + threadIdx.x; idx < total;
       idx += gridDim.x * blockDim.x) {
    int s    = idx / perMat;
    int r    = idx - s * perMat;
    int frag = r >> 9;
    int e    = r & 511;
    int nt   = frag >> 2;
    int kt   = frag & 3;
    int lane = e >> 4;
    int h    = e & 15;
    int n = nt * 16 + (lane & 15);
    int k = kt * 32 + ((lane >> 4) << 3) + (h < 8 ? h : h + 8);
    const float* w = args.w[s];
    unsigned short val = 0;
    if (w != nullptr && k < HIDN && n < HIDN) val = f2bf(w[k * HIDN + n]);
    wpack[idx] = val;
  }
}

__global__ void pack_fc_kernel(const float* fcw, unsigned short* fcpack) {
  const int total = 2000 * 4 * 512;  // ntiles * ktiles * halves/frag
  for (int idx = blockIdx.x * blockDim.x + threadIdx.x; idx < total;
       idx += gridDim.x * blockDim.x) {
    int frag = idx >> 9;
    int e    = idx & 511;
    int nt   = frag >> 2;
    int kt   = frag & 3;
    int lane = e >> 4;
    int h    = e & 15;
    int n = nt * 16 + (lane & 15);
    int k = kt * 32 + ((lane >> 4) << 3) + (h < 8 ? h : h + 8);
    unsigned short val = 0;
    if (k < HIDN) val = f2bf(fcw[k * VOC + n]);
    fcpack[idx] = val;
  }
}

// ---------------------------------------------------------------------------
// Embedding gather + layer-0 input projections (tiny; VALU is fine).
// Outputs time-major [t*32+b][100].
// ---------------------------------------------------------------------------
__global__ void embed_proj_kernel(const int* tokens, const float* emb,
                                  const float* wxr, const float* wxz, const float* wxh,
                                  const float* br, const float* bz, const float* bh,
                                  float* xr0, float* xz0, float* xh0) {
  __shared__ float e[EMBD];
  int bid = blockIdx.x;          // = t*32 + b
  int t = bid >> 5, b = bid & 31;
  int tid = threadIdx.x;
  if (tid < EMBD) {
    int tok = tokens[b * T_LEN + t];
    e[tid] = emb[tok * EMBD + tid];
  }
  __syncthreads();
  if (tid < HIDN) {
    float ar = br[tid], az = bz[tid], ah = bh[tid];
#pragma unroll 5
    for (int k = 0; k < EMBD; k++) {
      float ev = e[k];
      ar += ev * wxr[k * HIDN + tid];
      az += ev * wxz[k * HIDN + tid];
      ah += ev * wxh[k * HIDN + tid];
    }
    xr0[bid * HIDN + tid] = ar;
    xz0[bid * HIDN + tid] = az;
    xh0[bid * HIDN + tid] = ah;
  }
}

// ---------------------------------------------------------------------------
// Fused persistent kernel:
//   blocks 0..4   : pipelined GRU scan, one layer per block, weights in LDS
//   blocks 5..end : FC consumers, flag-gated per timestep, overlap the scan
// ---------------------------------------------------------------------------
struct Args {
  const unsigned short* wpack;
  const float* xr0; const float* xz0; const float* xh0;
  float* hbuf;               // [4][256][32][100] f32, layers 0..3 outputs
  unsigned short* hseq;      // [8192][128] bf16, layer-4 outputs (FC input)
  unsigned int* flags;       // [5][256]
  const unsigned short* fcpack;
  const float* fcb;
  float* out;
  const float* br[5]; const float* bz[5]; const float* bh[5];
};

__global__ void __launch_bounds__(256, 1)
gru_fused_kernel(Args args) {
  extern __shared__ char smemRaw[];
  const int tid  = threadIdx.x;
  const int wave = tid >> 5, lane = tid & 31;
  const v8f vzero = {0.f, 0.f, 0.f, 0.f, 0.f, 0.f, 0.f, 0.f};

  if (blockIdx.x < 5) {
    // ================= GRU scan pipeline =================
    unsigned short* wf   = (unsigned short*)smemRaw;       // 6*14336 halves
    unsigned short* hpB  = wf + 6 * 14336;                 // [32][128] bf16
    unsigned short* inpB = hpB + 32 * 128;                 // [32][128] bf16
    unsigned short* rhB  = inpB + 32 * 128;                // [32][128] bf16
    float* hpF  = (float*)(rhB + 32 * 128);                // [32][128] f32
    float* Rpre = hpF + 32 * 128;                          // [32][112] f32
    float* Zpre = Rpre + 32 * 112;                         // [32][112] f32

    const int layer = blockIdx.x;

    { // stage this layer's weight fragments into LDS once
      const uint4* src = (const uint4*)(args.wpack + (size_t)layer * 6 * 14336);
      uint4* dst = (uint4*)wf;
      for (int i = tid; i < 6 * 14336 / 8; i += 256) dst[i] = src[i];
    }
    for (int i = tid; i < 32 * 128; i += 256) { hpF[i] = 0.0f; hpB[i] = 0; }
    __syncthreads();

    const float* brl = args.br[layer];
    const float* bzl = args.bz[layer];
    const float* bhl = args.bh[layer];

    for (int t = 0; t < T_LEN; t++) {
      if (layer > 0) {
        const unsigned int* flg = &args.flags[(layer - 1) * T_LEN + t];
        while (__hip_atomic_load(flg, __ATOMIC_ACQUIRE, __HIP_MEMORY_SCOPE_AGENT) == 0u)
          __builtin_amdgcn_s_sleep(1);
        const float* src = args.hbuf + (((size_t)(layer - 1) * T_LEN + t) * BATCH) * HIDN;
        for (int i = tid; i < 32 * 128; i += 256) {
          int row = i >> 7, col = i & 127;
          inpB[i] = (col < HIDN) ? f2bf(src[row * HIDN + col]) : (unsigned short)0;
        }
      } else if (t + 1 < T_LEN && tid < 100) {
        // prefetch next timestep's precomputed gate inputs (global_prefetch_b8)
        __builtin_prefetch(&args.xr0[(size_t)(t + 1) * 3200 + tid * 32], 0, 1);
        __builtin_prefetch(&args.xz0[(size_t)(t + 1) * 3200 + tid * 32], 0, 1);
        __builtin_prefetch(&args.xh0[(size_t)(t + 1) * 3200 + tid * 32], 0, 1);
      }
      __syncthreads();

      // Phase 1: R and Z pre-activations (28 16x16 tiles over 8 waves)
      for (int tile = wave; tile < 28; tile += 8) {
        int g  = (tile >= 14) ? 1 : 0;
        int tt = tile - g * 14;
        int mt = tt / 7, nt = tt - mt * 7;
        int m0 = mt * 16;
        v8f acc = vzero;
        const unsigned short* wH = wf + (g ? 1 : 0) * 14336;   // W_hr / W_hz
#pragma unroll
        for (int k = 0; k < 4; k++)
          acc = wmma_bf16(load_afrag(hpB, m0, k, lane),
                          load_bfrag(wH, nt * 4 + k, lane), acc);
        if (layer > 0) {
          const unsigned short* wX = wf + (g ? 4 : 3) * 14336; // W_xr / W_xz
#pragma unroll
          for (int k = 0; k < 4; k++)
            acc = wmma_bf16(load_afrag(inpB, m0, k, lane),
                            load_bfrag(wX, nt * 4 + k, lane), acc);
        }
        float* dst = g ? Zpre : Rpre;
#pragma unroll
        for (int r = 0; r < 8; r++) {
          int m = m0 + r + ((lane >> 4) << 3);
          int n = nt * 16 + (lane & 15);
          dst[m * 112 + n] = acc[r];
        }
      }
      __syncthreads();

      // Phase 2: r = sigmoid(.), z = sigmoid(.), build (r*h) in bf16
      for (int i = tid; i < 32 * 128; i += 256) {
        int row = i >> 7, col = i & 127;
        unsigned short rh = 0;
        if (col < HIDN) {
          float ra = (layer == 0) ? args.xr0[(t * 32 + row) * HIDN + col] : brl[col];
          float za = (layer == 0) ? args.xz0[(t * 32 + row) * HIDN + col] : bzl[col];
          float rv = 1.0f / (1.0f + __expf(-(Rpre[row * 112 + col] + ra)));
          float zv = 1.0f / (1.0f + __expf(-(Zpre[row * 112 + col] + za)));
          Zpre[row * 112 + col] = zv;
          rh = f2bf(rv * hpF[i]);
        }
        rhB[i] = rh;
      }
      __syncthreads();

      // Phase 3: candidate pre-activation (14 tiles)
      for (int tile = wave; tile < 14; tile += 8) {
        int mt = tile / 7, nt = tile - mt * 7;
        int m0 = mt * 16;
        v8f acc = vzero;
        const unsigned short* wHH = wf + 2 * 14336;            // W_hh
#pragma unroll
        for (int k = 0; k < 4; k++)
          acc = wmma_bf16(load_afrag(rhB, m0, k, lane),
                          load_bfrag(wHH, nt * 4 + k, lane), acc);
        if (layer > 0) {
          const unsigned short* wXH = wf + 5 * 14336;          // W_xh
#pragma unroll
          for (int k = 0; k < 4; k++)
            acc = wmma_bf16(load_afrag(inpB, m0, k, lane),
                            load_bfrag(wXH, nt * 4 + k, lane), acc);
        }
#pragma unroll
        for (int r = 0; r < 8; r++) {
          int m = m0 + r + ((lane >> 4) << 3);
          int n = nt * 16 + (lane & 15);
          Rpre[m * 112 + n] = acc[r];
        }
      }
      __syncthreads();

      // Phase 4: h = z*tanh(.) + (1-z)*h ; export to next layer / FC input
      for (int i = tid; i < 32 * 128; i += 256) {
        int row = i >> 7, col = i & 127;
        if (col < HIDN) {
          float ha = (layer == 0) ? args.xh0[(t * 32 + row) * HIDN + col] : bhl[col];
          float ht = tanhf(Rpre[row * 112 + col] + ha);
          float zv = Zpre[row * 112 + col];
          float hn = zv * ht + (1.0f - zv) * hpF[i];
          hpF[i] = hn;
          unsigned short hb = f2bf(hn);
          hpB[i] = hb;
          if (layer < 4)
            args.hbuf[(((size_t)layer * T_LEN + t) * BATCH + row) * HIDN + col] = hn;
          else
            args.hseq[(size_t)(t * 32 + row) * 128 + col] = hb;
        }
      }
      __threadfence();
      __syncthreads();
      if (tid == 0)
        __hip_atomic_store(&args.flags[layer * T_LEN + t], 1u,
                           __ATOMIC_RELEASE, __HIP_MEMORY_SCOPE_AGENT);
    }
  } else {
    // ================= FC consumers (overlap the scan) =================
    // Job = (t, nb): 32x640 f32 output tile; needs only layer-4 h at timestep t.
    float* stage = (float*)smemRaw;    // [32][640] f32 staging for coalesced stores
    const int f = blockIdx.x - 5;

    for (int job = f; job < NJOBS; job += NFC) {
      int t  = job / 50;
      int nb = job - t * 50;

      // prefetch this job's packed weights while (possibly) waiting
      __builtin_prefetch(&args.fcpack[((size_t)(nb * 40 + wave * 5) * 4) * 512 + lane * 64], 0, 1);

      const unsigned int* flg = &args.flags[4 * T_LEN + t];
      while (__hip_atomic_load(flg, __ATOMIC_ACQUIRE, __HIP_MEMORY_SCOPE_AGENT) == 0u)
        __builtin_amdgcn_s_sleep(4);

      int m0     = t * 32;
      int ntBase = nb * 40 + wave * 5;   // 16-col tile index
      v8f acc[2][5];
#pragma unroll
      for (int mt = 0; mt < 2; mt++)
#pragma unroll
        for (int j = 0; j < 5; j++) acc[mt][j] = vzero;

#pragma unroll
      for (int k = 0; k < 4; k++) {
        v16bf a0 = load_afrag(args.hseq, m0, k, lane);
        v16bf a1 = load_afrag(args.hseq, m0 + 16, k, lane);
#pragma unroll
        for (int j = 0; j < 5; j++) {
          v16bf b = load_bfrag(args.fcpack, (ntBase + j) * 4 + k, lane);
          acc[0][j] = wmma_bf16(a0, b, acc[0][j]);
          acc[1][j] = wmma_bf16(a1, b, acc[1][j]);
        }
      }

      // deposit into LDS stage (+bias), then fully-coalesced float4 stores
#pragma unroll
      for (int j = 0; j < 5; j++) {
        int nl   = (wave * 5 + j) * 16 + (lane & 15);
        float bv = args.fcb[nb * 640 + nl];
#pragma unroll
        for (int mt = 0; mt < 2; mt++) {
#pragma unroll
          for (int r = 0; r < 8; r++) {
            int ml = mt * 16 + r + ((lane >> 4) << 3);
            stage[ml * 640 + nl] = acc[mt][j][r] + bv;
          }
        }
      }
      __syncthreads();

      for (int i = tid; i < 32 * 160; i += 256) {     // 160 float4 per row
        int row = i / 160, c4 = i - row * 160;
        int m = m0 + row;
        int bb = m & 31, tt = m >> 5;                 // row index = t*32 + b
        *(float4*)(&args.out[((size_t)bb * T_LEN + tt) * VOC + nb * 640 + c4 * 4]) =
            *(const float4*)(&stage[row * 640 + c4 * 4]);
      }
      __syncthreads();
    }
  }
}

// ---------------------------------------------------------------------------
extern "C" void kernel_launch(void* const* d_in, const int* in_sizes, int n_in,
                              void* d_out, int out_size, void* d_ws, size_t ws_size,
                              hipStream_t stream) {
  (void)in_sizes; (void)n_in; (void)out_size; (void)ws_size;
  // d_in order: 0 inputs, 1 emb, 2-6 W_hr, 7-11 W_xr, 12-16 b_r, 17-21 W_hz,
  // 22-26 W_xz, 27-31 b_z, 32-36 W_hh, 37-41 W_xh, 42-46 b_h, 47 fc_w, 48 fc_b
  const int*   tokens = (const int*)d_in[0];
  const float* emb    = (const float*)d_in[1];
  const float* fcw    = (const float*)d_in[47];
  const float* fcb    = (const float*)d_in[48];

  // Workspace carve (bytes)
  char* ws = (char*)d_ws;
  size_t off = 0;
  unsigned int*   flags  = (unsigned int*)(ws + off);   off += 8192;
  unsigned short* hseq   = (unsigned short*)(ws + off); off += (size_t)8192 * 128 * 2;
  float*          xr0    = (float*)(ws + off);          off += (size_t)8192 * HIDN * 4;
  float*          xz0    = (float*)(ws + off);          off += (size_t)8192 * HIDN * 4;
  float*          xh0    = (float*)(ws + off);          off += (size_t)8192 * HIDN * 4;
  float*          hbuf   = (float*)(ws + off);          off += (size_t)4 * T_LEN * BATCH * HIDN * 4;
  unsigned short* wpack  = (unsigned short*)(ws + off); off += (size_t)30 * 14336 * 2;
  unsigned short* fcpack = (unsigned short*)(ws + off); off += (size_t)2000 * 4 * 512 * 2;

  // Zero flags + hseq padding (graph-capturable memset node)
  hipMemsetAsync(d_ws, 0, 8192 + (size_t)8192 * 128 * 2, stream);

  // Pack recurrent weights (slots: l*6 + {W_hr,W_hz,W_hh,W_xr,W_xz,W_xh})
  PackArgs pa;
  for (int l = 0; l < 5; l++) {
    pa.w[l * 6 + 0] = (const float*)d_in[2 + l];
    pa.w[l * 6 + 1] = (const float*)d_in[17 + l];
    pa.w[l * 6 + 2] = (const float*)d_in[32 + l];
    pa.w[l * 6 + 3] = (l == 0) ? nullptr : (const float*)d_in[7 + l];
    pa.w[l * 6 + 4] = (l == 0) ? nullptr : (const float*)d_in[22 + l];
    pa.w[l * 6 + 5] = (l == 0) ? nullptr : (const float*)d_in[37 + l];
  }
  pack_gru_kernel<<<1680, 256, 0, stream>>>(pa, wpack);
  pack_fc_kernel<<<16000, 256, 0, stream>>>(fcw, fcpack);

  embed_proj_kernel<<<8192, 128, 0, stream>>>(
      tokens, emb,
      (const float*)d_in[7], (const float*)d_in[22], (const float*)d_in[37],
      (const float*)d_in[12], (const float*)d_in[27], (const float*)d_in[42],
      xr0, xz0, xh0);

  Args a;
  a.wpack = wpack; a.xr0 = xr0; a.xz0 = xz0; a.xh0 = xh0;
  a.hbuf = hbuf; a.hseq = hseq; a.flags = flags;
  a.fcpack = fcpack; a.fcb = fcb; a.out = (float*)d_out;
  for (int l = 0; l < 5; l++) {
    a.br[l] = (const float*)d_in[12 + l];
    a.bz[l] = (const float*)d_in[27 + l];
    a.bh[l] = (const float*)d_in[42 + l];
  }
  const int SMEM = 6 * 14336 * 2      // weight fragments (bf16)
                 + 3 * 32 * 128 * 2   // hpB, inpB, rhB
                 + 32 * 128 * 4       // hpF
                 + 2 * 32 * 112 * 4;  // Rpre, Zpre  => 241,664 B (>= 80KB FC stage)
  hipFuncSetAttribute((const void*)gru_fused_kernel,
                      hipFuncAttributeMaxDynamicSharedMemorySize, SMEM);
  gru_fused_kernel<<<5 + NFC, 256, SMEM, stream>>>(a);
}
